// mLSTMClassifier_51213190037939
// MI455X (gfx1250) — matrix operand order repaired and checked by reading, not compiled, not run on hardware
//
#include <hip/hip_runtime.h>
#include <hip/hip_bf16.h>
#include <math.h>

typedef _Float16 half_t;
typedef __attribute__((ext_vector_type(16))) _Float16 v16h;
typedef __attribute__((ext_vector_type(8)))  _Float16 v8h;
typedef __attribute__((ext_vector_type(8)))  float    v8f;

#define B_      32
#define S_      512
#define E_      512
#define INNER_  1024
#define BS_     (B_ * S_)
#define NH_     4
#define DH_     256

// ---------------- block reductions (blockDim == 256) ----------------
__device__ __forceinline__ float block_reduce_sum(float v, float* sred) {
    int t = threadIdx.x;
    sred[t] = v; __syncthreads();
    for (int off = 128; off > 0; off >>= 1) {
        if (t < off) sred[t] += sred[t + off];
        __syncthreads();
    }
    float r = sred[0]; __syncthreads();
    return r;
}
__device__ __forceinline__ float block_reduce_max(float v, float* sred) {
    int t = threadIdx.x;
    sred[t] = v; __syncthreads();
    for (int off = 128; off > 0; off >>= 1) {
        if (t < off) sred[t] = fmaxf(sred[t], sred[t + off]);
        __syncthreads();
    }
    float r = sred[0]; __syncthreads();
    return r;
}

// ---------------- pack W (N x K row-major, B[k][n] = W[n][k]) into WMMA B-fragment layout ----
// out[((kt*NT + nt)*32 + lane)*16 + e] = W[row_off + nt*16 + (lane&15)][kt*32 + (e>>3)*16 + (lane>>4)*8 + (e&7)]
__global__ __launch_bounds__(256) void pack_b_kernel(const float* __restrict__ W,
                                                     half_t* __restrict__ out,
                                                     int N, int K, int row_off) {
    int o = blockIdx.x * 256 + threadIdx.x;
    if (o >= N * K) return;
    int e    = o & 15;
    int lane = (o >> 4) & 31;
    int NT   = N >> 4;
    int nt   = (o >> 9) % NT;
    int kt   = o / (512 * NT);
    int n = nt * 16 + (lane & 15);
    int k = kt * 32 + (e >> 3) * 16 + (lane >> 4) * 8 + (e & 7);
    out[o] = (half_t)W[(size_t)(row_off + n) * K + k];
}

__device__ __forceinline__ v16h load_a_frag(const half_t* arow) {
    v8h lo = *(const v8h*)(arow);
    v8h hi = *(const v8h*)(arow + 16);
    v16h a;
    #pragma unroll
    for (int i = 0; i < 8; ++i) { a[i] = lo[i]; a[8 + i] = hi[i]; }
    return a;
}

// ---------------- generic 1-tile WMMA GEMM (used for the tiny M=32 GEMMs) ----------------
__global__ __launch_bounds__(256) void gemm_wmma_f16(const half_t* __restrict__ A,
                                                     const half_t* __restrict__ Bp,
                                                     half_t* __restrict__ C,
                                                     int M, int N, int K) {
    const int lane = threadIdx.x & 31;
    const int wave = threadIdx.x >> 5;
    const int mt_cnt = M >> 4, nt_cnt = N >> 4, kt_cnt = K >> 5;
    int tile = blockIdx.x * 8 + wave;
    if (tile >= mt_cnt * nt_cnt) return;
    int mt = tile % mt_cnt;
    int nt = tile / mt_cnt;
    const int row = lane & 15, khalf = lane >> 4;

    v8f acc = {};
    const half_t* arow = A + (size_t)(mt * 16 + row) * K + khalf * 8;
    const size_t  bstride = (size_t)nt_cnt * 512;
    const half_t* bptr = Bp + (size_t)nt * 512 + (size_t)lane * 16;

    for (int kt = 0; kt < kt_cnt; ++kt) {
        v16h a = load_a_frag(arow);
        v16h b = *(const v16h*)(bptr);
        acc = __builtin_amdgcn_wmma_f32_16x16x32_f16(false, a, false, b,
                                                     (short)0, acc, false, false);
        arow += 32;
        bptr += bstride;
    }
    half_t* crow = C + (size_t)(mt * 16 + khalf * 8) * N + nt * 16 + (lane & 15);
    #pragma unroll
    for (int r = 0; r < 8; ++r) crow[(size_t)r * N] = (half_t)acc[r];
}

// ---------------- register-blocked WMMA GEMM: each wave does 4 mt x 2 nt tiles -----------
// Requires M % 64 == 0, N % 32 == 0, K % 32 == 0. 8 waves/block share one nt-pair (B in L1).
__global__ __launch_bounds__(256) void gemm_wmma_f16_4x2(const half_t* __restrict__ A,
                                                         const half_t* __restrict__ Bp,
                                                         half_t* __restrict__ C,
                                                         int M, int N, int K) {
    const int lane = threadIdx.x & 31;
    const int wave = threadIdx.x >> 5;
    const int MQ = M >> 6, NQ = N >> 5;          // quad-m, pair-n job grid
    const int nt_cnt = N >> 4, kt_cnt = K >> 5;
    int job = blockIdx.x * 8 + wave;
    if (job >= MQ * NQ) return;
    int mq = job % MQ;
    int nq = job / MQ;
    const int row = lane & 15, khalf = lane >> 4;

    v8f acc[4][2];
    #pragma unroll
    for (int i = 0; i < 4; ++i)
        #pragma unroll
        for (int j = 0; j < 2; ++j) acc[i][j] = (v8f){};

    const half_t* arow = A + (size_t)(mq * 64 + row) * K + khalf * 8;
    const size_t  astep = (size_t)16 * K;         // next mt within the quad
    const size_t  bstride = (size_t)nt_cnt * 512; // next kt
    const half_t* bptr = Bp + (size_t)(nq * 2) * 512 + (size_t)lane * 16;

    for (int kt = 0; kt < kt_cnt; ++kt) {
        if (kt + 1 < kt_cnt) {                    // gfx1250 global_prefetch_b8
            __builtin_prefetch(arow + 32, 0, 0);
            __builtin_prefetch(bptr + bstride, 0, 0);
        }
        v16h a0 = load_a_frag(arow);
        v16h a1 = load_a_frag(arow + astep);
        v16h a2 = load_a_frag(arow + 2 * astep);
        v16h a3 = load_a_frag(arow + 3 * astep);
        v16h b0 = *(const v16h*)(bptr);
        v16h b1 = *(const v16h*)(bptr + 512);
        acc[0][0] = __builtin_amdgcn_wmma_f32_16x16x32_f16(false, a0, false, b0, (short)0, acc[0][0], false, false);
        acc[1][0] = __builtin_amdgcn_wmma_f32_16x16x32_f16(false, a1, false, b0, (short)0, acc[1][0], false, false);
        acc[2][0] = __builtin_amdgcn_wmma_f32_16x16x32_f16(false, a2, false, b0, (short)0, acc[2][0], false, false);
        acc[3][0] = __builtin_amdgcn_wmma_f32_16x16x32_f16(false, a3, false, b0, (short)0, acc[3][0], false, false);
        acc[0][1] = __builtin_amdgcn_wmma_f32_16x16x32_f16(false, a0, false, b1, (short)0, acc[0][1], false, false);
        acc[1][1] = __builtin_amdgcn_wmma_f32_16x16x32_f16(false, a1, false, b1, (short)0, acc[1][1], false, false);
        acc[2][1] = __builtin_amdgcn_wmma_f32_16x16x32_f16(false, a2, false, b1, (short)0, acc[2][1], false, false);
        acc[3][1] = __builtin_amdgcn_wmma_f32_16x16x32_f16(false, a3, false, b1, (short)0, acc[3][1], false, false);
        arow += 32;
        bptr += bstride;
    }
    #pragma unroll
    for (int i = 0; i < 4; ++i) {
        #pragma unroll
        for (int j = 0; j < 2; ++j) {
            half_t* crow = C + (size_t)(mq * 64 + i * 16 + khalf * 8) * N
                             + (nq * 2 + j) * 16 + (lane & 15);
            #pragma unroll
            for (int r = 0; r < 8; ++r) crow[(size_t)r * N] = (half_t)acc[i][j][r];
        }
    }
}

// ---------------- embedding gather + LN1 -> h(f16); stash last-token residual/h -----------
__global__ __launch_bounds__(256) void embed_ln_kernel(const int* __restrict__ tokens,
                                                       const float* __restrict__ emb,
                                                       const float* __restrict__ ln1w,
                                                       half_t* __restrict__ h16,
                                                       half_t* __restrict__ hlast16,
                                                       float* __restrict__ reslast) {
    __shared__ float sred[256];
    int bs = blockIdx.x;
    int s = bs % S_, b = bs / S_;
    int tok = tokens[bs];
    int e0 = threadIdx.x, e1 = threadIdx.x + 256;
    float x0 = (tok == 0) ? 0.f : emb[(size_t)tok * E_ + e0];
    float x1 = (tok == 0) ? 0.f : emb[(size_t)tok * E_ + e1];
    float mean = block_reduce_sum(x0 + x1, sred) * (1.f / 512.f);
    float d0 = x0 - mean, d1 = x1 - mean;
    float var = block_reduce_sum(d0 * d0 + d1 * d1, sred) * (1.f / 512.f);
    float inv = rsqrtf(var + 1e-5f);
    float h0 = d0 * inv * ln1w[e0];
    float h1 = d1 * inv * ln1w[e1];
    h16[(size_t)bs * E_ + e0] = (half_t)h0;
    h16[(size_t)bs * E_ + e1] = (half_t)h1;
    if (s == S_ - 1) {
        reslast[b * E_ + e0] = x0;  reslast[b * E_ + e1] = x1;
        hlast16[b * E_ + e0] = (half_t)h0;  hlast16[b * E_ + e1] = (half_t)h1;
    }
}

// ---------------- causal depthwise conv(k=4) + bias + SiLU ----------------
__global__ __launch_bounds__(256) void conv_silu_kernel(const half_t* __restrict__ xm,
                                                        const float* __restrict__ cw,
                                                        const float* __restrict__ cb,
                                                        half_t* __restrict__ xa) {
    size_t idx = (size_t)blockIdx.x * 256 + threadIdx.x;   // over B*S*INNER
    int c = (int)(idx % INNER_);
    int s = (int)((idx / INNER_) % S_);
    size_t bbase = idx - (size_t)c - (size_t)s * INNER_;
    float acc = cb[c];
    #pragma unroll
    for (int j = 0; j < 4; ++j) {
        int ss = s - 3 + j;
        if (ss >= 0) acc += (float)xm[bbase + (size_t)ss * INNER_ + c] * cw[c * 4 + j];
    }
    float sg = 1.f / (1.f + __expf(-acc));
    xa[idx] = (half_t)(acc * sg);
}

// ---------------- blockwise 4x4 q/k/v + gate dot products ----------------
__global__ __launch_bounds__(256) void qkv_gates_kernel(const half_t* __restrict__ xa,
                                                        const half_t* __restrict__ xm,
                                                        const float* __restrict__ qw,
                                                        const float* __restrict__ kw,
                                                        const float* __restrict__ vw,
                                                        const float* __restrict__ igw,
                                                        const float* __restrict__ igb,
                                                        const float* __restrict__ fgw,
                                                        const float* __restrict__ fgb,
                                                        half_t* __restrict__ k16,
                                                        half_t* __restrict__ v16,
                                                        float* __restrict__ qlast,
                                                        float* __restrict__ ig,
                                                        float* __restrict__ fg) {
    __shared__ float sred[256];
    int bs = blockIdx.x;
    int s = bs % S_, b = bs / S_;
    int nb = threadIdx.x;                    // 256 blocks of 4 channels
    size_t base = (size_t)bs * INNER_ + nb * 4;
    float xa4[4], xm4[4];
    #pragma unroll
    for (int i = 0; i < 4; ++i) { xa4[i] = (float)xa[base + i]; xm4[i] = (float)xm[base + i]; }
    float q4[4], k4[4], v4[4];
    #pragma unroll
    for (int o = 0; o < 4; ++o) {
        float aq = 0.f, ak = 0.f, av = 0.f;
        #pragma unroll
        for (int i = 0; i < 4; ++i) {
            int wi = (nb * 4 + o) * 4 + i;
            aq += xa4[i] * qw[wi];
            ak += xa4[i] * kw[wi];
            av += xm4[i] * vw[wi];
        }
        q4[o] = aq; k4[o] = ak; v4[o] = av;
    }
    #pragma unroll
    for (int o = 0; o < 4; ++o) { k16[base + o] = (half_t)k4[o]; v16[base + o] = (half_t)v4[o]; }
    if (s == S_ - 1) {
        #pragma unroll
        for (int o = 0; o < 4; ++o) qlast[b * INNER_ + nb * 4 + o] = q4[o];
    }
    float part[8];
    #pragma unroll
    for (int h = 0; h < NH_; ++h) {
        float pi = 0.f, pf = 0.f;
        #pragma unroll
        for (int o = 0; o < 4; ++o) {
            int c = nb * 4 + o;
            pi += q4[o] * igw[h * 3072 + c] + k4[o] * igw[h * 3072 + 1024 + c]
                + v4[o] * igw[h * 3072 + 2048 + c];
            pf += q4[o] * fgw[h * 3072 + c] + k4[o] * fgw[h * 3072 + 1024 + c]
                + v4[o] * fgw[h * 3072 + 2048 + c];
        }
        part[h] = pi; part[4 + h] = pf;
    }
    for (int t = 0; t < 8; ++t) {
        float tot = block_reduce_sum(part[t], sred);
        if (threadIdx.x == 0) {
            if (t < 4) ig[(size_t)bs * 4 + t]       = tot + igb[t];
            else       fg[(size_t)bs * 4 + (t - 4)] = tot + fgb[t - 4];
        }
        __syncthreads();
    }
}

// ---------------- mLSTM row s=S-1 only, head-LN, skip+gate -> hs_last ----------------
__global__ __launch_bounds__(256) void attn_last_kernel(const half_t* __restrict__ k16,
                                                        const half_t* __restrict__ v16,
                                                        const float* __restrict__ ig,
                                                        const float* __restrict__ fg,
                                                        const float* __restrict__ qlast,
                                                        const half_t* __restrict__ zlast,
                                                        const half_t* __restrict__ xa,
                                                        const float* __restrict__ onorm,
                                                        const float* __restrict__ skipw,
                                                        half_t* __restrict__ hslast) {
    __shared__ float sc[512];    // cumsum(log_sigmoid(fg))
    __shared__ float scm[512];   // logd -> cm
    __shared__ float sq[256];    // q row
    __shared__ float sred[256];
    int b = blockIdx.x >> 2, h = blockIdx.x & 3;
    int t = threadIdx.x;

    sq[t] = qlast[b * INNER_ + h * DH_ + t];
    for (int i = t; i < 512; i += 256) {
        float f = fg[((size_t)b * S_ + i) * 4 + h];
        sc[i] = fminf(f, 0.f) - log1pf(__expf(-fabsf(f)));   // log_sigmoid
    }
    __syncthreads();
    if (t == 0) { float run = 0.f; for (int i = 0; i < 512; ++i) { run += sc[i]; sc[i] = run; } }
    __syncthreads();
    float cS = sc[511];

    float lmax = -3.4e38f;
    for (int i = t; i < 512; i += 256) {
        float ld = cS - sc[i] + ig[((size_t)b * S_ + i) * 4 + h];
        scm[i] = ld;
        lmax = fmaxf(lmax, ld);
    }
    float maxd = block_reduce_max(lmax, sred);

    float csum_part = 0.f;
    for (int i = t; i < 512; i += 256) {
        const v8h* kp8 = (const v8h*)(k16 + ((size_t)b * S_ + i) * INNER_ + h * DH_);
        float qk = 0.f;
        for (int d8 = 0; d8 < DH_ / 8; ++d8) {
            v8h kv = kp8[d8];
            #pragma unroll
            for (int u = 0; u < 8; ++u) qk += sq[d8 * 8 + u] * (float)kv[u];
        }
        qk *= 0.0625f;                                  // DH^-0.5 = 1/16
        float cm = qk * __expf(scm[i] - maxd);
        scm[i] = cm;
        csum_part += cm;
    }
    float csum = block_reduce_sum(csum_part, sred);
    float nrm = fmaxf(fabsf(csum), __expf(-maxd)) + 1e-6f;

    int dd = t;                                         // output dim 0..255
    const half_t* vp = v16 + (size_t)b * S_ * INNER_ + h * DH_ + dd;
    float acc = 0.f;
    for (int i = 0; i < 512; ++i) acc += scm[i] * (float)vp[(size_t)i * INNER_];
    float o = acc / nrm;

    float mean = block_reduce_sum(o, sred) * (1.f / 256.f);
    float dev = o - mean;
    float var = block_reduce_sum(dev * dev, sred) * (1.f / 256.f);
    float hn = dev * rsqrtf(var + 1e-5f);
    int c = h * DH_ + dd;
    float hnw = hn * onorm[c];
    float xal = (float)xa[(((size_t)b * S_) + (S_ - 1)) * INNER_ + c];
    float z = (float)zlast[b * INNER_ + c];
    float sz = z / (1.f + __expf(-z));
    hslast[b * INNER_ + c] = (half_t)((hnw + skipw[c] * xal) * sz);
}

// ---------------- final: LN(res+y) * ln_post, dot fc_w, sigmoid ----------------
__global__ __launch_bounds__(256) void final_kernel(const float* __restrict__ reslast,
                                                    const half_t* __restrict__ ylast,
                                                    const float* __restrict__ lnpw,
                                                    const float* __restrict__ fcw,
                                                    const float* __restrict__ fcb,
                                                    float* __restrict__ out) {
    __shared__ float sred[256];
    int b = blockIdx.x, t = threadIdx.x;
    float x0 = reslast[b * E_ + t]       + (float)ylast[b * E_ + t];
    float x1 = reslast[b * E_ + 256 + t] + (float)ylast[b * E_ + 256 + t];
    float mean = block_reduce_sum(x0 + x1, sred) * (1.f / 512.f);
    float d0 = x0 - mean, d1 = x1 - mean;
    float var = block_reduce_sum(d0 * d0 + d1 * d1, sred) * (1.f / 512.f);
    float inv = rsqrtf(var + 1e-5f);
    float xn0 = d0 * inv * lnpw[t], xn1 = d1 * inv * lnpw[t + 256];
    float dot = block_reduce_sum(xn0 * fcw[t] + xn1 * fcw[t + 256], sred);
    if (t == 0) out[b] = 1.f / (1.f + __expf(-(dot + fcb[0])));
}

// ---------------- host launcher ----------------
extern "C" void kernel_launch(void* const* d_in, const int* in_sizes, int n_in,
                              void* d_out, int out_size, void* d_ws, size_t ws_size,
                              hipStream_t stream) {
    const int*   tokens = (const int*)  d_in[0];
    const float* emb    = (const float*)d_in[1];
    const float* ln1w   = (const float*)d_in[2];
    const float* convw  = (const float*)d_in[3];
    const float* convb  = (const float*)d_in[4];
    const float* wup    = (const float*)d_in[5];
    const float* qw     = (const float*)d_in[6];
    const float* kw     = (const float*)d_in[7];
    const float* vw     = (const float*)d_in[8];
    const float* igw    = (const float*)d_in[9];
    const float* igb    = (const float*)d_in[10];
    const float* fgw    = (const float*)d_in[11];
    const float* fgb    = (const float*)d_in[12];
    const float* onormw = (const float*)d_in[13];
    const float* skipw  = (const float*)d_in[14];
    const float* wdown  = (const float*)d_in[15];
    const float* lnpw   = (const float*)d_in[16];
    const float* fcw    = (const float*)d_in[17];
    const float* fcb    = (const float*)d_in[18];
    float* out = (float*)d_out;

    char* p = (char*)d_ws;
    auto take = [&](size_t bytes) -> char* {
        char* r = p; p += (bytes + 255) & ~(size_t)255; return r;
    };
    half_t* h16     = (half_t*)take((size_t)BS_ * E_ * 2);        // 16 MB
    half_t* hlast16 = (half_t*)take((size_t)B_ * E_ * 2);
    float*  reslast = (float*) take((size_t)B_ * E_ * 4);
    half_t* wpm     = (half_t*)take((size_t)INNER_ * E_ * 2);     // 1 MB
    half_t* wpz     = (half_t*)take((size_t)INNER_ * E_ * 2);
    half_t* wdp     = (half_t*)take((size_t)E_ * INNER_ * 2);
    half_t* xm16    = (half_t*)take((size_t)BS_ * INNER_ * 2);    // 32 MB
    half_t* xa16    = (half_t*)take((size_t)BS_ * INNER_ * 2);
    half_t* k16     = (half_t*)take((size_t)BS_ * INNER_ * 2);
    half_t* v16b    = (half_t*)take((size_t)BS_ * INNER_ * 2);
    half_t* zlast   = (half_t*)take((size_t)B_ * INNER_ * 2);
    float*  qlast   = (float*) take((size_t)B_ * INNER_ * 4);
    float*  igbuf   = (float*) take((size_t)BS_ * NH_ * 4);
    float*  fgbuf   = (float*) take((size_t)BS_ * NH_ * 4);
    half_t* hslast  = (half_t*)take((size_t)B_ * INNER_ * 2);
    half_t* ylast   = (half_t*)take((size_t)B_ * E_ * 2);
    (void)ws_size; (void)in_sizes; (void)n_in; (void)out_size;

    // pack B operands into WMMA fragment layout (re-done each call; deterministic)
    pack_b_kernel<<<(INNER_ * E_ + 255) / 256, 256, 0, stream>>>(wup,   wpm, INNER_, E_, 0);
    pack_b_kernel<<<(INNER_ * E_ + 255) / 256, 256, 0, stream>>>(wup,   wpz, INNER_, E_, INNER_);
    pack_b_kernel<<<(E_ * INNER_ + 255) / 256, 256, 0, stream>>>(wdown, wdp, E_, INNER_, 0);

    embed_ln_kernel<<<BS_, 256, 0, stream>>>(tokens, emb, ln1w, h16, hlast16, reslast);

    // x_m = h @ w_up[:1024].T  (big GEMM: 16384x1024x512, register-blocked 4x2)
    {
        int jobs = (BS_ / 64) * (INNER_ / 32);    // 256 * 32 = 8192 wave jobs
        gemm_wmma_f16_4x2<<<(jobs + 7) / 8, 256, 0, stream>>>(h16, wpm, xm16, BS_, INNER_, E_);
    }
    // z_last = h_last @ w_up[1024:].T  (32x1024x512)
    {
        int tiles = (B_ / 16) * (INNER_ / 16);
        gemm_wmma_f16<<<(tiles + 7) / 8, 256, 0, stream>>>(hlast16, wpz, zlast, B_, INNER_, E_);
    }

    conv_silu_kernel<<<(BS_ * INNER_) / 256, 256, 0, stream>>>(xm16, convw, convb, xa16);

    qkv_gates_kernel<<<BS_, 256, 0, stream>>>(xa16, xm16, qw, kw, vw, igw, igb, fgw, fgb,
                                              k16, v16b, qlast, igbuf, fgbuf);

    attn_last_kernel<<<B_ * NH_, 256, 0, stream>>>(k16, v16b, igbuf, fgbuf, qlast, zlast,
                                                   xa16, onormw, skipw, hslast);

    // y_last = hs_last @ w_down.T  (32x512x1024)
    {
        int tiles = (B_ / 16) * (E_ / 16);
        gemm_wmma_f16<<<(tiles + 7) / 8, 256, 0, stream>>>(hslast, wdp, ylast, B_, E_, INNER_);
    }

    final_kernel<<<B_, 256, 0, stream>>>(reslast, ylast, lnpw, fcw, fcb, out);
}
